// BaseAttender_37323265802782
// MI455X (gfx1250) — matrix operand order repaired and checked by reading, not compiled, not run on hardware
//
#include <hip/hip_runtime.h>

// ---------------------------------------------------------------------------
// Masked attention + output projection for MI455X (gfx1250, wave32, WMMA+TDM).
// B=16, NQ=1024, NK=2048, D=512, V=512, O=256.  Memory-bound (~390 MB, int32
// mask dominates at 134 MB); all matmuls on v_wmma_f32_16x16x32_bf16 with f32
// accumulation; V tiles staged into double-buffered LDS by the Tensor Data
// Mover (tensor_load_to_lds, TENSORcnt) so DMA overlaps WMMA issue.
// ---------------------------------------------------------------------------

typedef __bf16 bf16;
typedef bf16  v8bf  __attribute__((ext_vector_type(8)));
typedef bf16  v16bf __attribute__((ext_vector_type(16)));
typedef float v8f   __attribute__((ext_vector_type(8)));
typedef unsigned int u32x4 __attribute__((ext_vector_type(4)));
typedef int i32x4 __attribute__((ext_vector_type(4)));
typedef int i32x8 __attribute__((ext_vector_type(8)));

#define CAT16(lo, hi) __builtin_shufflevector(lo, hi, 0,1,2,3,4,5,6,7,8,9,10,11,12,13,14,15)

static __device__ __forceinline__ v16bf load_frag16(const bf16* p) {
  // 16 contiguous bf16 (per-lane K run k = 16*h + s: this file's contiguous order for both A and B)
  v8bf lo = *(const v8bf*)(p);
  v8bf hi = *(const v8bf*)(p + 8);
  return CAT16(lo, hi);
}
static __device__ __forceinline__ v16bf load_frag_split(const bf16* p0, const bf16* p1) {
  // A-matrix fragment: two 8-element runs (K 0..7 / 16..23 or 8..15 / 24..31)
  v8bf lo = *(const v8bf*)(p0);
  v8bf hi = *(const v8bf*)(p1);
  return CAT16(lo, hi);
}

// Generic pointer -> raw LDS byte offset (addrspace(3) pointers are the LDS
// byte offset on AMDGPU).  Used only inside function bodies, never in a
// static initializer (lld cannot relocate addrspacecast of LDS symbols).
typedef __attribute__((address_space(3))) const void lds_cv;
static __device__ __forceinline__ unsigned lds_addr_of(const void* p) {
  return (unsigned)(uintptr_t)(lds_cv*)p;
}

// ---------------------------------------------------------------------------
// TDM: DMA one [512 v][32 k] bf16 tile from global V^T (row stride 2048 elems)
// into LDS with an 8-element pad per row (LDS row pitch = 40 bf16 = 80 B).
// Descriptor packing per CDNA5 ISA 08_async_tensor 8.3/8.4.
// ---------------------------------------------------------------------------
static __device__ __forceinline__ void tdm_load_vtile(const bf16* gsrc, unsigned lds_byte_off) {
  unsigned long long ga = (unsigned long long)(uintptr_t)gsrc;
  u32x4 g0;
  g0[0] = 1u;                                             // count=1 (valid), user mode
  g0[1] = lds_byte_off;                                   // lds_addr [63:32]
  g0[2] = (unsigned)ga;                                   // global_addr[31:0]
  g0[3] = ((unsigned)(ga >> 32) & 0x01ffffffu)            // global_addr[56:32]
        | (2u << 30);                                     // type=2 ("image")
  i32x8 g1;
  g1[0] = (1 << 16)            // data_size = 1 -> 2 bytes
        | (1 << 20)            // pad_enable
        | (3 << 22)            // pad_interval: 16 DWORDs (= one 32-elem row)
        | (3 << 25);           // pad_amount: 4 DWORDs (= 8 bf16 pad)
  g1[1] = (int)(2048u << 16);  // tensor_dim0 = 2048 (k extent)
  g1[2] = (int)(512u << 16);   // tensor_dim1 = 512  (v extent)
  g1[3] = (int)(32u << 16);    // tile_dim0 = 32 (k)
  g1[4] = 512;                 // tile_dim1 = 512 (v), tile_dim2 = 0
  g1[5] = 2048;                // tensor_dim0_stride[31:0] = 2048 elems
  g1[6] = 0;
  g1[7] = 0;
  i32x4 z4 = {0, 0, 0, 0};
#if __clang_major__ >= 23
  i32x8 z8 = {0, 0, 0, 0, 0, 0, 0, 0};
  __builtin_amdgcn_tensor_load_to_lds(g0, g1, z4, z4, z8, 0);
#else
  __builtin_amdgcn_tensor_load_to_lds(g0, g1, z4, z4, 0);
#endif
}

// ---------------------------------------------------------------------------
// fp32 -> bf16 bulk convert (4 elems/thread)
// ---------------------------------------------------------------------------
__global__ void cvt4_kernel(const float* __restrict__ in, bf16* __restrict__ o, int n) {
  int i = (blockIdx.x * blockDim.x + threadIdx.x) * 4;
  if (i + 3 < n) {
    float4 f = *(const float4*)(in + i);
    o[i + 0] = (bf16)f.x;
    o[i + 1] = (bf16)f.y;
    o[i + 2] = (bf16)f.z;
    o[i + 3] = (bf16)f.w;
  }
}

// ---------------------------------------------------------------------------
// V [b][k][v] fp32  ->  V^T [b][v][k] bf16 (LDS-tiled 32x32 transpose)
// Grid: 16 batches * 64 ktiles * 16 vtiles = 16384 blocks of 256 threads.
// ---------------------------------------------------------------------------
__global__ __launch_bounds__(256)
void transpose_v_kernel(const float* __restrict__ Vf, bf16* __restrict__ Vt) {
  __shared__ bf16 tile[32][33];
  const int b  = blockIdx.x >> 10;
  const int t  = blockIdx.x & 1023;
  const int kt = t >> 4;          // 0..63
  const int vt = t & 15;          // 0..15
  const int tx = threadIdx.x & 31;
  const int ty = threadIdx.x >> 5;  // 0..7
  const float* src = Vf + ((size_t)b * 2048 + kt * 32) * 512 + vt * 32;
  #pragma unroll
  for (int i = 0; i < 32; i += 8)
    tile[ty + i][tx] = (bf16)src[(size_t)(ty + i) * 512 + tx];
  __syncthreads();
  bf16* dst = Vt + ((size_t)b * 512 + vt * 32) * 2048 + kt * 32;
  #pragma unroll
  for (int i = 0; i < 32; i += 8)
    dst[(size_t)(ty + i) * 2048 + tx] = tile[tx][ty + i];
}

// ---------------------------------------------------------------------------
// Flash attention: each wave owns a 16-row q tile, iterates 32-wide k tiles.
// Grid: 128 blocks (16 batches x 8 q-slabs of 128 rows), 256 threads = 8 waves.
// Dynamic LDS: Vt[2][512*40] bf16 (TDM double buffer) + Pbuf[8*16*40] bf16.
// ---------------------------------------------------------------------------
#define VT_STRIDE 40
#define VT_BYTES  (512 * VT_STRIDE * 2)          // 40960 per buffer
#define ATTN_SMEM (2 * VT_BYTES + 8 * 16 * VT_STRIDE * 2)  // 92160

__global__ __launch_bounds__(256, 1)
void attn_kernel(const bf16* __restrict__ Qbf, const bf16* __restrict__ Kbf,
                 const bf16* __restrict__ Vtg, const int* __restrict__ mask,
                 bf16* __restrict__ ctx)
{
  constexpr int NQc = 1024, NKc = 2048, Dc = 512, VDc = 512;
  extern __shared__ __align__(16) char smem_raw[];
  const unsigned lds0 = lds_addr_of(smem_raw);

  const int wave = threadIdx.x >> 5;
  const int lane = threadIdx.x & 31;
  const int lo16 = lane & 15;
  const int hi   = lane >> 4;          // half-wave select
  const int aoff = hi ? 8 : 0;         // A-frag K offset
  const int boff = hi ? 16 : 0;        // B-frag K offset
  const int roff = hi ? 8 : 0;         // C-layout row offset

  const int b    = blockIdx.x >> 3;
  const int q0   = (blockIdx.x & 7) * 128 + wave * 16;

  const bf16* qrow  = Qbf + ((size_t)b * NQc + q0 + lo16) * Dc;
  const bf16* kbase = Kbf + (size_t)b * NKc * Dc;
  const bf16* vtg_b = Vtg + (size_t)b * VDc * NKc;   // V^T for this batch
  const int*  mbase = mask + ((size_t)b * NQc + q0 + roff) * NKc;

  const v8f zero8 = {0.f,0.f,0.f,0.f,0.f,0.f,0.f,0.f};
  v8f acc[32];                         // 16 x 512 f32 context accumulator
  #pragma unroll
  for (int i = 0; i < 32; ++i) acc[i] = zero8;

  float mrow[8], lrow[8];
  #pragma unroll
  for (int r = 0; r < 8; ++r) { mrow[r] = -1e9f; lrow[r] = 0.f; }

  const float scale = 0.044194173824159216f;  // 1/sqrt(512)

  // Prime the pipeline: TDM copy of tile 0 into buffer 0 (one op per block).
  if (wave == 0) tdm_load_vtile(vtg_b, lds0);

  #pragma unroll 1
  for (int kt = 0; kt < NKc / 32; ++kt) {
    const int k0 = kt * 32;

    // Wait for this tile's DMA, make it visible to all waves, then kick off
    // the DMA for the next tile into the other buffer (overlaps compute).
    if (wave == 0) __builtin_amdgcn_s_wait_tensorcnt(0);
    __syncthreads();
    if (wave == 0 && kt + 1 < NKc / 32)
      tdm_load_vtile(vtg_b + (kt + 1) * 32, lds0 + ((unsigned)((kt + 1) & 1)) * VT_BYTES);
    // Current tile's LDS buffer (computed pointer -- no static initializer).
    const bf16* Vtb = (const bf16*)(smem_raw + (size_t)(kt & 1) * VT_BYTES);
    bf16* Pbuf = (bf16*)(smem_raw + 2 * VT_BYTES);

    // ---- S = Q * K^T  (two 16x16 tiles, K=512 -> 32 WMMAs) ----
    v8f S0 = zero8, S1 = zero8;
    const bf16* krow0 = kbase + (size_t)(k0 + lo16) * Dc;
    const bf16* krow1 = krow0 + (size_t)16 * Dc;
    #pragma unroll
    for (int d0 = 0; d0 < Dc; d0 += 32) {
      v16bf a  = load_frag16(qrow + d0 + boff);
      v16bf b0 = load_frag16(krow0 + d0 + boff);
      v16bf b1 = load_frag16(krow1 + d0 + boff);
      S0 = __builtin_amdgcn_wmma_f32_16x16x32_bf16(false, a, false, b0, (short)0, S0, false, false);
      S1 = __builtin_amdgcn_wmma_f32_16x16x32_bf16(false, a, false, b1, (short)0, S1, false, false);
    }

    // ---- mask + online softmax (row stats per C-layout row) ----
    float p0[8], p1[8], alpha[8];
    #pragma unroll
    for (int r = 0; r < 8; ++r) {
      const int* mr = mbase + (size_t)r * NKc + k0;
      __builtin_prefetch(mr + 32, 0, 0);   // next k-tile of the mask stream
      float s0 = (mr[lo16]      != 0) ? S0[r] * scale : -1e9f;
      float s1 = (mr[16 + lo16] != 0) ? S1[r] * scale : -1e9f;
      float t = fmaxf(s0, s1);
      t = fmaxf(t, __shfl_xor(t, 1, 16));
      t = fmaxf(t, __shfl_xor(t, 2, 16));
      t = fmaxf(t, __shfl_xor(t, 4, 16));
      t = fmaxf(t, __shfl_xor(t, 8, 16));
      float mn = fmaxf(mrow[r], t);
      float al = __expf(mrow[r] - mn);
      float e0 = __expf(s0 - mn);
      float e1 = __expf(s1 - mn);
      float rs = e0 + e1;
      rs += __shfl_xor(rs, 1, 16);
      rs += __shfl_xor(rs, 2, 16);
      rs += __shfl_xor(rs, 4, 16);
      rs += __shfl_xor(rs, 8, 16);
      lrow[r]  = lrow[r] * al + rs;
      mrow[r]  = mn;
      alpha[r] = al;
      p0[r] = e0;
      p1[r] = e1;
    }

    // ---- rescale running accumulator ----
    #pragma unroll
    for (int vc = 0; vc < 32; ++vc)
      #pragma unroll
      for (int r = 0; r < 8; ++r)
        acc[vc][r] *= alpha[r];

    // ---- P: C-layout -> A-layout via wave-private LDS slice ----
    bf16* pb = &Pbuf[wave * 16 * VT_STRIDE];
    #pragma unroll
    for (int r = 0; r < 8; ++r) {
      pb[(r + roff) * VT_STRIDE + lo16]      = (bf16)p0[r];
      pb[(r + roff) * VT_STRIDE + 16 + lo16] = (bf16)p1[r];
    }
    asm volatile("s_wait_dscnt 0" ::: "memory");   // wave-internal cross-lane LDS RAW

    const bf16* prow = &Pbuf[(wave * 16 + lo16) * VT_STRIDE];
    v16bf ap = load_frag16(prow + boff);

    // ---- acc += P * Vtile  (32 column chunks -> 32 WMMAs) ----
    #pragma unroll
    for (int vc = 0; vc < 32; ++vc) {
      const bf16* vr = &Vtb[(size_t)(vc * 16 + lo16) * VT_STRIDE + boff];
      v16bf bv = load_frag16(vr);
      acc[vc] = __builtin_amdgcn_wmma_f32_16x16x32_bf16(false, ap, false, bv, (short)0, acc[vc], false, false);
    }
    // No trailing barrier needed: the top-of-loop wait+barrier orders the next
    // TDM write (other buffer) and the next tile's reads.
  }

  // ---- normalize and emit bf16 context [B*NQ, 512] ----
  float linv[8];
  #pragma unroll
  for (int r = 0; r < 8; ++r) linv[r] = 1.0f / lrow[r];
  bf16* crow = ctx + ((size_t)b * NQc + q0 + roff) * VDc + lo16;
  #pragma unroll
  for (int vc = 0; vc < 32; ++vc)
    #pragma unroll
    for (int r = 0; r < 8; ++r)
      crow[(size_t)r * VDc + vc * 16] = (bf16)(acc[vc][r] * linv[r]);
}

// ---------------------------------------------------------------------------
// Projection: out[16384,256] = ctx[16384,512] * W^T[512,256] + bias.
// One 16x16 output tile per wave (16 WMMAs), 8 waves/block, 2048 blocks.
// ---------------------------------------------------------------------------
__global__ __launch_bounds__(256)
void proj_kernel(const bf16* __restrict__ ctx, const bf16* __restrict__ Wbf,
                 const float* __restrict__ bias, float* __restrict__ out)
{
  const int wave = threadIdx.x >> 5;
  const int lane = threadIdx.x & 31;
  const int lo16 = lane & 15;
  const int hi   = lane >> 4;
  const int aoff = hi ? 8 : 0;
  const int boff = hi ? 16 : 0;
  const int roff = hi ? 8 : 0;

  const int tile = blockIdx.x * 8 + wave;   // 16384 tiles total
  const int mt = tile >> 4;                 // 0..1023 row tile
  const int nt = tile & 15;                 // 0..15  col tile

  const bf16* arow = ctx + (size_t)(mt * 16 + lo16) * 512;
  const bf16* brow = Wbf + (size_t)(nt * 16 + lo16) * 512;  // W row o = contiguous v

  v8f c = {0.f,0.f,0.f,0.f,0.f,0.f,0.f,0.f};
  #pragma unroll
  for (int v0 = 0; v0 < 512; v0 += 32) {
    v16bf a  = load_frag16(arow + v0 + boff);
    v16bf bm = load_frag16(brow + v0 + boff);
    c = __builtin_amdgcn_wmma_f32_16x16x32_bf16(false, a, false, bm, (short)0, c, false, false);
  }

  const float bs = bias[nt * 16 + lo16];
  float* orow = out + (size_t)(mt * 16 + roff) * 256 + nt * 16 + lo16;
  #pragma unroll
  for (int r = 0; r < 8; ++r) orow[(size_t)r * 256] = c[r] + bs;
}

// ---------------------------------------------------------------------------
extern "C" void kernel_launch(void* const* d_in, const int* in_sizes, int n_in,
                              void* d_out, int out_size, void* d_ws, size_t ws_size,
                              hipStream_t stream) {
  (void)in_sizes; (void)n_in; (void)out_size; (void)ws_size;

  const float* keys    = (const float*)d_in[0];   // [16,2048,512]
  const float* queries = (const float*)d_in[1];   // [16,1024,512]
  const float* values  = (const float*)d_in[2];   // [16,2048,512]
  const int*   maskp   = (const int*)d_in[3];     // [16,1024,2048]
  const float* Wr      = (const float*)d_in[4];   // [256,512]
  const float* br      = (const float*)d_in[5];   // [256]
  float* out = (float*)d_out;                     // [16,1024,256] f32

  // workspace layout (96.25 MiB total)
  char* ws = (char*)d_ws;
  bf16* Qbf = (bf16*)(ws);                                   // 16 MiB
  bf16* Kbf = (bf16*)(ws + (size_t)16 * 1024 * 1024);        // 32 MiB
  bf16* Vtg = (bf16*)(ws + (size_t)48 * 1024 * 1024);        // 32 MiB (V^T bf16)
  bf16* ctx = (bf16*)(ws + (size_t)80 * 1024 * 1024);        // 16 MiB
  bf16* Wbf = (bf16*)(ws + (size_t)96 * 1024 * 1024);        // 256 KiB

  const int nQ = 16 * 1024 * 512;   // 8388608
  const int nK = 16 * 2048 * 512;   // 16777216
  const int nW = 256 * 512;         // 131072

  cvt4_kernel<<<(nQ / 4 + 255) / 256, 256, 0, stream>>>(queries, Qbf, nQ);
  cvt4_kernel<<<(nK / 4 + 255) / 256, 256, 0, stream>>>(keys,    Kbf, nK);
  cvt4_kernel<<<(nW / 4 + 255) / 256, 256, 0, stream>>>(Wr,      Wbf, nW);
  transpose_v_kernel<<<16384, 256, 0, stream>>>(values, Vtg);

  (void)hipFuncSetAttribute((const void*)attn_kernel,
                            hipFuncAttributeMaxDynamicSharedMemorySize, ATTN_SMEM);
  attn_kernel<<<128, 256, ATTN_SMEM, stream>>>(Qbf, Kbf, Vtg, maskp, ctx);
  proj_kernel<<<2048, 256, 0, stream>>>(ctx, Wbf, br, out);
}
